// CustomLinear_82197084111396
// MI455X (gfx1250) — matrix operand order repaired and checked by reading, not compiled, not run on hardware
//
#include <hip/hip_runtime.h>

// BFP-quantized GEMM for MI455X (gfx1250, wave32, WMMA).
//   out = quant16(x) @ quant16(weight)^T + bias,  M=N=K=4096
// Dequantized BFP values (8-bit mantissa * 2^e) are exactly representable in
// fp16, so we quantize both operands to fp16 in d_ws and run an exact
// v_wmma_f32_16x16x32_f16 GEMM with f32 accumulation.

typedef _Float16 half_t;
typedef __attribute__((ext_vector_type(16))) _Float16 v16h;
typedef __attribute__((ext_vector_type(8)))  _Float16 v8h;
typedef __attribute__((ext_vector_type(8)))  float    v8f;

#define M_DIM 4096
#define N_DIM 4096
#define K_DIM 4096
#define GROUP 16

#define BM 128
#define BN 128
#define BK 32
#define LDA 40   // padded LDS row stride in halfs (80B -> conflict-free b128 rows)
#define LDB 40

// ---------------------------------------------------------------------------
// Pass 1: block-floating-point quantize-dequantize fp32 -> fp16.
// One thread per group of 16 consecutive elements along K.
// ---------------------------------------------------------------------------
__global__ __launch_bounds__(256)
void bfp_quant_kernel(const float* __restrict__ in, half_t* __restrict__ out,
                      int ngroups)
{
    int g = blockIdx.x * blockDim.x + threadIdx.x;
    if (g >= ngroups) return;

    const float4* p = (const float4*)(in + (size_t)g * GROUP);
    float x[GROUP];
#pragma unroll
    for (int c = 0; c < 4; ++c) {
        float4 t = p[c];
        x[c * 4 + 0] = t.x; x[c * 4 + 1] = t.y;
        x[c * 4 + 2] = t.z; x[c * 4 + 3] = t.w;
    }

    float maxabs = 0.0f;
#pragma unroll
    for (int i = 0; i < GROUP; ++i) maxabs = fmaxf(maxabs, fabsf(x[i]));

    v8h lo, hi;
    if (maxabs == 0.0f) {
#pragma unroll
        for (int i = 0; i < 8; ++i) { lo[i] = (_Float16)0.0f; hi[i] = (_Float16)0.0f; }
    } else {
        int e;
        (void)frexpf(fmaxf(maxabs, 1e-38f), &e);
        e -= 1;                                  // floor(log2(maxabs))
        float scale = ldexpf(1.0f, e - 6);       // 2^(e - (MANT-2))
        float inv   = ldexpf(1.0f, 6 - e);
#pragma unroll
        for (int i = 0; i < GROUP; ++i) {
            float q = rintf(x[i] * inv);                    // round half-to-even
            q = fminf(127.0f, fmaxf(-128.0f, q)) * scale;   // clip & dequant
            if (i < 8) lo[i] = (_Float16)q; else hi[i - 8] = (_Float16)q;
        }
    }

    v8h* o = (v8h*)(out + (size_t)g * GROUP);
    o[0] = lo;
    o[1] = hi;
}

// ---------------------------------------------------------------------------
// Pass 2: C[M,N] = A[M,K] * B[N,K]^T + bias, fp16 operands, f32 accum via WMMA.
// 256 threads = 8 waves (wave32), 2x4 wave grid, 128x128 block tile, BK=32.
// Each wave: 4 (M) x 2 (N) tiles of 16x16 -> 8 v_wmma per K-step.
// ---------------------------------------------------------------------------
__global__ __launch_bounds__(256)
void bfp_gemm_kernel(const half_t* __restrict__ A, const half_t* __restrict__ B,
                     const float* __restrict__ bias, float* __restrict__ C)
{
    __shared__ half_t sA[BM * LDA];
    __shared__ half_t sB[BN * LDB];

    const int tid  = threadIdx.x;
    const int lane = tid & 31;
    const int wave = tid >> 5;      // 0..7
    const int wm   = wave >> 2;     // 0..1 : 64-row slab
    const int wn   = wave & 3;      // 0..3 : 32-col slab

    const int m0 = blockIdx.y * BM;
    const int n0 = blockIdx.x * BN;

    v8f acc[4][2] = {};

    // Global->LDS staging: thread t covers row t>>2 (and +64), 8-half chunk t&3.
    const int lrow   = tid >> 2;
    const int lchunk = (tid & 3) * 8;

    const int hs = lane >> 4;   // half-wave select (0: lanes 0-15, 1: lanes 16-31)
    const int r  = lane & 15;

    for (int k0 = 0; k0 < K_DIM; k0 += BK) {
        v8h a0 = *(const v8h*)(A + (size_t)(m0 + lrow)      * K_DIM + k0 + lchunk);
        v8h a1 = *(const v8h*)(A + (size_t)(m0 + lrow + 64) * K_DIM + k0 + lchunk);
        v8h b0 = *(const v8h*)(B + (size_t)(n0 + lrow)      * K_DIM + k0 + lchunk);
        v8h b1 = *(const v8h*)(B + (size_t)(n0 + lrow + 64) * K_DIM + k0 + lchunk);

        __syncthreads();   // previous iteration done reading LDS
        *(v8h*)(sA + (size_t)lrow        * LDA + lchunk) = a0;
        *(v8h*)(sA + (size_t)(lrow + 64) * LDA + lchunk) = a1;
        *(v8h*)(sB + (size_t)lrow        * LDB + lchunk) = b0;
        *(v8h*)(sB + (size_t)(lrow + 64) * LDB + lchunk) = b1;
        __syncthreads();

        // A fragment (16x32 f16): lanes 0-15 hold K {0..7, 16..23} of row M=r,
        // lanes 16-31 hold K {8..15, 24..31}  -> two contiguous b128 reads.
        union Frag { v16h v; v8h h[2]; };
        Frag af[4], bf[2];
        const int klo_a = hs * 8;
#pragma unroll
        for (int mi = 0; mi < 4; ++mi) {
            const half_t* base = sA + (size_t)(wm * 64 + mi * 16 + r) * LDA;
            af[mi].h[0] = *(const v8h*)(base + klo_a);
            af[mi].h[1] = *(const v8h*)(base + klo_a + 16);
        }
        // B fragment (32x16 f16): lanes 0-15 hold K {0..7, 16..23} of column N=r,
        // lanes 16-31 hold K {8..15, 24..31} -> two contiguous b128 reads.
        const int klo_b = hs * 8;
#pragma unroll
        for (int ni = 0; ni < 2; ++ni) {
            const half_t* base = sB + (size_t)(wn * 32 + ni * 16 + r) * LDB;
            bf[ni].h[0] = *(const v8h*)(base + klo_b);
            bf[ni].h[1] = *(const v8h*)(base + klo_b + 16);
        }

#pragma unroll
        for (int mi = 0; mi < 4; ++mi)
#pragma unroll
            for (int ni = 0; ni < 2; ++ni)
                acc[mi][ni] = __builtin_amdgcn_wmma_f32_16x16x32_f16(
                    /*neg_a=*/false, af[mi].v, /*neg_b=*/false, bf[ni].v,
                    /*c_mod=*/(short)0, acc[mi][ni],
                    /*reuse_a=*/false, /*reuse_b=*/false);
    }

    // Epilogue. C/D layout: VGPR i -> M = i + 8*hs, N = lane%16.
#pragma unroll
    for (int ni = 0; ni < 2; ++ni) {
        const int n = n0 + wn * 32 + ni * 16 + r;
        const float bv = bias[n];
#pragma unroll
        for (int mi = 0; mi < 4; ++mi) {
            const int mbase = m0 + wm * 64 + mi * 16 + hs * 8;
#pragma unroll
            for (int i = 0; i < 8; ++i)
                C[(size_t)(mbase + i) * N_DIM + n] = acc[mi][ni][i] + bv;
        }
    }
}

// ---------------------------------------------------------------------------
extern "C" void kernel_launch(void* const* d_in, const int* in_sizes, int n_in,
                              void* d_out, int out_size, void* d_ws, size_t ws_size,
                              hipStream_t stream)
{
    const float* x    = (const float*)d_in[0];   // [M, K] fp32
    const float* w    = (const float*)d_in[1];   // [N, K] fp32
    const float* bias = (const float*)d_in[2];   // [N]    fp32
    float* out = (float*)d_out;                  // [M, N] fp32

    // Workspace: quantized fp16 copies of x and weight (33.5 MB each).
    half_t* xq = (half_t*)d_ws;
    half_t* wq = xq + (size_t)M_DIM * K_DIM;

    const int ngroups = (M_DIM * K_DIM) / GROUP;   // 1,048,576 per matrix
    bfp_quant_kernel<<<ngroups / 256, 256, 0, stream>>>(x, xq, ngroups);
    bfp_quant_kernel<<<ngroups / 256, 256, 0, stream>>>(w, wq, ngroups);

    dim3 grid(N_DIM / BN, M_DIM / BM);   // 32 x 32 blocks
    bfp_gemm_kernel<<<grid, 256, 0, stream>>>(xq, wq, bias, out);
}
